// MLPResNetBlock_44066364457544
// MI455X (gfx1250) — matrix operand order repaired and checked by reading, not compiled, not run on hardware
//
#include <hip/hip_runtime.h>
#include <hip/hip_bf16.h>
#include <math.h>

#define DIMC 1024
#define NHEAD 8
#define HD 128
#define BB 8
#define TT 1024
#define KT 512
#define KA 128                 // h_a(64) + p(64)
#define S_ALL (TT + KA + KT)   // 1664
#define S_TSK_OFF (TT + KA)    // 1152 (multiple of 32: chunks never straddle)

typedef __bf16 bf16;
typedef __attribute__((ext_vector_type(16))) __bf16 bf16x16;
typedef __attribute__((ext_vector_type(8)))  __bf16 bf16x8;
typedef __attribute__((ext_vector_type(8)))  float  f32x8;

union FragU { bf16x16 v; bf16x8 h[2]; };

#if __has_builtin(__builtin_amdgcn_global_load_async_to_lds_b128)
#define USE_ASYNC_COPY 1
typedef int v4i __attribute__((vector_size(16)));
typedef __attribute__((address_space(1))) v4i gas_v4i;  // global AS
typedef __attribute__((address_space(3))) v4i las_v4i;  // LDS AS
#else
#define USE_ASYNC_COPY 0
#endif

// 16-byte global -> LDS copy: async (ASYNCcnt-tracked DMA) when available.
__device__ inline void cp16_g2l(const bf16* g, bf16* l) {
#if USE_ASYNC_COPY
  __builtin_amdgcn_global_load_async_to_lds_b128(
      (gas_v4i*)(unsigned long long)g,
      (las_v4i*)(unsigned int)(unsigned long long)l, 0, 0);
#else
  *(bf16x8*)l = *(const bf16x8*)g;
#endif
}

template <int N>
__device__ inline void async_wait() {
#if USE_ASYNC_COPY
#if __has_builtin(__builtin_amdgcn_s_wait_asynccnt)
  __builtin_amdgcn_s_wait_asynccnt(N);
#else
  asm volatile("s_wait_asynccnt %0" ::"i"(N) : "memory");
#endif
#endif
}

__device__ inline f32x8 zero8() {
  f32x8 z = {0.f,0.f,0.f,0.f,0.f,0.f,0.f,0.f};
  return z;
}

// A-matrix fragment (16x32 bf16): lane holds row (lane&15),
// K-runs [koff, koff+8) and [koff+16, koff+24), koff = 8*(lane>=16).
__device__ inline bf16x16 load_frag_a(const bf16* p0, int stride, int lane) {
  int r    = lane & 15;
  int koff = (lane >> 4) << 3;
  const bf16* p = p0 + (size_t)r * stride + koff;
  FragU u;
  u.h[0] = *(const bf16x8*)(p);
  u.h[1] = *(const bf16x8*)(p + 16);
  return u.v;
}

// B-matrix fragment (32x16 bf16): lane holds column (lane&15),
// K-runs [koff, koff+8) and [koff+16, koff+24), koff = 8*(lane>=16).
// For C = A * W^T with W row-major [N,K], column n of B == row n of W.
__device__ inline bf16x16 load_frag_b(const bf16* p0, int stride, int lane) {
  int r     = lane & 15;
  int kbase = (lane >> 4) << 3;
  const bf16* p = p0 + (size_t)r * stride + kbase;
  FragU u;
  u.h[0] = *(const bf16x8*)(p);
  u.h[1] = *(const bf16x8*)(p + 16);
  return u.v;
}

__device__ inline f32x8 wmma_bf16(bf16x16 a, bf16x16 b, f32x8 c) {
  return __builtin_amdgcn_wmma_f32_16x16x32_bf16(
      false, a, false, b, (short)0, c, false, false);
}

// ---------------------------------------------------------------------------
// fp32 -> bf16 convert with batched row remap (for concat into K/V buffers)
// ---------------------------------------------------------------------------
__global__ void cvt_bf16_k(const float* __restrict__ in, bf16* __restrict__ out,
                           long n_pb, long out_stride_pb, long seg_elems,
                           long total) {
  for (long i = (long)blockIdx.x * blockDim.x + threadIdx.x; i < total;
       i += (long)gridDim.x * blockDim.x) {
    long b   = i / n_pb;
    long rem = i - b * n_pb;
    out[b * out_stride_pb + seg_elems + rem] = (bf16)in[i];
  }
}

// ---------------------------------------------------------------------------
// GEMM: C[M,1024] = A[M,1024] (bf16) * W[1024,1024]^T (bf16) + bias
// MODE 0: bf16 out;  MODE 1: f32 out + residual add;  MODE 2: f32 out + relu
// grid: (8, rows_pb/128, B); block: 256 (8 waves); wave tile 32x64.
// Double-buffered async LDS staging of A/W 128x32 tiles (ASYNCcnt pipelined).
// ---------------------------------------------------------------------------
#define LDS_STRIDE 48   // 96B rows: 16B aligned, spreads banks

template <int MODE>
__global__ __launch_bounds__(256)
void gemm_bf16_k(const bf16* __restrict__ A, const bf16* __restrict__ W,
                 const float* __restrict__ bias, int a_rows_pb,
                 int out_rows_pb, int out_seg,
                 bf16* __restrict__ outB, float* __restrict__ outF,
                 const float* __restrict__ resid) {
  __shared__ bf16 smA[2][128 * LDS_STRIDE];
  __shared__ bf16 smW[2][128 * LDS_STRIDE];

  const int tid  = threadIdx.x;
  const int lane = tid & 31;
  const int w    = tid >> 5;
  const int wm   = w >> 1;            // 0..3
  const int wn   = w & 1;             // 0..1
  const int z    = blockIdx.z;

  const int m0 = blockIdx.y * 128;
  const int n0 = blockIdx.x * 128;

  const bf16* Abase = A + (size_t)z * a_rows_pb * DIMC + (size_t)m0 * DIMC;
  const bf16* Wbase = W + (size_t)n0 * DIMC;

  // stage one 128x32 tile (A or W) into LDS: 2 x 16B per thread
  auto stage = [&](const bf16* gbase, bf16* lds, int k0) {
#pragma unroll
    for (int j = 0; j < 2; j++) {
      int c   = tid + 256 * j;        // 0..511 chunks of 16B
      int row = c >> 2, q = c & 3;    // 4 chunks per 32-elem row
      cp16_g2l(gbase + (size_t)row * DIMC + k0 + q * 8,
               lds + row * LDS_STRIDE + q * 8);
    }
  };

  f32x8 acc[2][4];
#pragma unroll
  for (int i = 0; i < 2; i++)
#pragma unroll
    for (int j = 0; j < 4; j++) acc[i][j] = zero8();

  stage(Abase, smA[0], 0);
  stage(Wbase, smW[0], 0);
  int cur = 0;

  for (int k0 = 0; k0 < DIMC; k0 += 32) {
    if (k0 + 32 < DIMC) {
      stage(Abase, smA[cur ^ 1], k0 + 32);
      stage(Wbase, smW[cur ^ 1], k0 + 32);
      async_wait<4>();   // current stage done; next stage still in flight
    } else {
      async_wait<0>();
    }
    __syncthreads();

    const bf16* sa = smA[cur] + (wm * 32) * LDS_STRIDE;
    const bf16* sw = smW[cur] + (wn * 64) * LDS_STRIDE;
    bf16x16 a0 = load_frag_a(sa, LDS_STRIDE, lane);
    bf16x16 a1 = load_frag_a(sa + 16 * LDS_STRIDE, LDS_STRIDE, lane);
    bf16x16 bfr[4];
#pragma unroll
    for (int j = 0; j < 4; j++)
      bfr[j] = load_frag_b(sw + j * 16 * LDS_STRIDE, LDS_STRIDE, lane);
#pragma unroll
    for (int j = 0; j < 4; j++) {
      acc[0][j] = wmma_bf16(a0, bfr[j], acc[0][j]);
      acc[1][j] = wmma_bf16(a1, bfr[j], acc[1][j]);
    }
    __syncthreads();
    cur ^= 1;
  }

  const int nn   = lane & 15;
  const int mrow = (lane >> 4) << 3;
#pragma unroll
  for (int mi = 0; mi < 2; mi++)
#pragma unroll
    for (int nj = 0; nj < 4; nj++) {
      int ncol = n0 + wn * 64 + nj * 16 + nn;
      float bv = bias[ncol];
      size_t rbase =
          ((size_t)z * out_rows_pb + out_seg + m0 + wm * 32 + mi * 16 + mrow);
#pragma unroll
      for (int i = 0; i < 8; i++) {
        size_t idx = (rbase + i) * DIMC + ncol;
        float  v   = acc[mi][nj][i] + bv;
        if (MODE == 0)      outB[idx] = (bf16)v;
        else if (MODE == 1) outF[idx] = v + resid[idx];
        else                outF[idx] = v > 0.f ? v : 0.f;
      }
    }
}

// ---------------------------------------------------------------------------
// Flash attention over S_all = 1664, TRANSPOSED formulation:
//   S^T = K * Q^T   (softmax dimension s lives within a lane)
//   O^T = V^T * P^T (output d x m; packed b128 stores)
// grid: (T/128, H, B); block 256 (8 waves); wave owns 16 query cols, D=128.
// ---------------------------------------------------------------------------
__global__ __launch_bounds__(256)
void attn_k(const bf16* __restrict__ Q, const bf16* __restrict__ Km,
            const bf16* __restrict__ Vm, const float* __restrict__ gate,
            bf16* __restrict__ Ao) {
  __shared__ bf16 KTs[32 * 128];        // K chunk, row-major [s][d]
  __shared__ bf16 VT[128 * 48];         // V^T chunk [d][s], padded rows
  __shared__ bf16 PT[8 * 16 * 32];      // per-wave P^T tile stored [m][s]

  const int tid  = threadIdx.x;
  const int lane = tid & 31;
  const int w    = tid >> 5;
  const int b    = blockIdx.z;
  const int h    = blockIdx.y;
  const int t0   = blockIdx.x * 128 + w * 16;

  const int mcol = lane & 15;           // query column owned by this lane
  const int half = lane >> 4;           // 0 or 1
  const int soff = half << 3;           // s offset of this lane's C rows

  const float ratio_g = tanhf(gate[0]);
  const float invs    = 0.08838834764831845f;  // 1/sqrt(128)

  const bf16* Qb = Q  + ((size_t)b * TT)    * DIMC + (size_t)h * HD;
  const bf16* Kb = Km + ((size_t)b * S_ALL) * DIMC + (size_t)h * HD;
  const bf16* Vb = Vm + ((size_t)b * S_ALL) * DIMC + (size_t)h * HD;

  // Q as B-matrix fragments (column m = Q row t0+m), 4 d-chunks of 32
  bf16x16 qf[4];
#pragma unroll
  for (int dc = 0; dc < 4; dc++)
    qf[dc] = load_frag_b(Qb + (size_t)t0 * DIMC + dc * 32, DIMC, lane);

  f32x8 o[8];                           // O^T accum: o[dt][i] = (d=dt*16+soff+i, m=mcol)
#pragma unroll
  for (int dt = 0; dt < 8; dt++) o[dt] = zero8();
  float m_run = -1.0e30f, l_run = 0.f;  // per-lane scalars (column m)

  bf16* ptw = PT + w * 16 * 32;         // this wave's P^T tile [m][s], stride 32

  for (int s0 = 0; s0 < S_ALL; s0 += 32) {
    // async K staging: 32 rows x 128 d, row-major (2 x 16B per thread)
#pragma unroll
    for (int j = 0; j < 2; j++) {
      int c   = tid + 256 * j;        // 0..511 chunks
      int row = c >> 4, q = c & 15;   // 16 chunks per 128-elem row
      cp16_g2l(Kb + (size_t)(s0 + row) * DIMC + q * 8,
               KTs + row * 128 + q * 8);
    }
    // V^T staging (transpose; element-wise): VT[d][s] = V[s0+s][d]
    for (int idx = tid; idx < 32 * 128; idx += 256) {
      int s = idx >> 7, d = idx & 127;
      VT[d * 48 + s] = Vb[(size_t)(s0 + s) * DIMC + d];
    }
    async_wait<0>();
    __syncthreads();

    // S^T: two 16x16 tiles (s-rows 0..15 and 16..31), A = K rows from LDS
    bf16x16 kfr[8];
#pragma unroll
    for (int dc = 0; dc < 4; dc++) {
      kfr[2 * dc + 0] = load_frag_a(KTs + dc * 32, 128, lane);
      kfr[2 * dc + 1] = load_frag_a(KTs + 16 * 128 + dc * 32, 128, lane);
    }
    f32x8 sc0 = zero8(), sc1 = zero8();
#pragma unroll
    for (int dc = 0; dc < 4; dc++) {
      sc0 = wmma_bf16(kfr[2 * dc + 0], qf[dc], sc0);
      sc1 = wmma_bf16(kfr[2 * dc + 1], qf[dc], sc1);
    }
    const float segs = (s0 >= S_TSK_OFF) ? ratio_g * invs : invs;

    // ---- online softmax along s: 16 values in-lane + 16 in partner lane ----
    float x0[8], x1[8];
    float mx = -1.0e30f;
#pragma unroll
    for (int i = 0; i < 8; i++) {
      x0[i] = sc0[i] * segs;            // s = soff + i
      x1[i] = sc1[i] * segs;            // s = 16 + soff + i
      mx = fmaxf(mx, fmaxf(x0[i], x1[i]));
    }
    mx = fmaxf(mx, __shfl_xor(mx, 16, 32));   // combine with partner half
    const float nm   = fmaxf(m_run, mx);
    const float corr = __expf(m_run - nm);
    bf16x8 pk0, pk1;
    float rs = 0.f;
#pragma unroll
    for (int i = 0; i < 8; i++) {
      float p0 = __expf(x0[i] - nm);
      float p1 = __expf(x1[i] - nm);
      rs += p0 + p1;
      pk0[i] = (bf16)p0;
      pk1[i] = (bf16)p1;
    }
    rs += __shfl_xor(rs, 16, 32);
    l_run = l_run * corr + rs;
    m_run = nm;
#pragma unroll
    for (int dt = 0; dt < 8; dt++) o[dt] *= corr;

    // P^T tile: row m = mcol, packed 8-element stores at s = soff, 16+soff
    *(bf16x8*)(ptw + mcol * 32 + soff)      = pk0;
    *(bf16x8*)(ptw + mcol * 32 + 16 + soff) = pk1;

    // O^T += V^T * P^T : A = V^T rows (LDS), B = P^T columns (LDS)
    bf16x16 pf = load_frag_b(ptw, 32, lane);
    bf16x16 vfr[8];
#pragma unroll
    for (int dt = 0; dt < 8; dt++)
      vfr[dt] = load_frag_a(VT + (dt * 16) * 48, 48, lane);
#pragma unroll
    for (int dt = 0; dt < 8; dt++)
      o[dt] = wmma_bf16(vfr[dt], pf, o[dt]);
    __syncthreads();
  }

  // normalize + packed output: lane stores 8 consecutive d per tile (b128)
  const float rl = 1.f / l_run;
  bf16* orow = Ao + ((size_t)b * TT + t0 + mcol) * DIMC + h * HD + soff;
#pragma unroll
  for (int dt = 0; dt < 8; dt++) {
    bf16x8 ov;
#pragma unroll
    for (int i = 0; i < 8; i++) ov[i] = (bf16)(o[dt][i] * rl);
    *(bf16x8*)(orow + dt * 16) = ov;
  }
}

// ---------------------------------------------------------------------------
// LayerNorm over DIMC per row, write bf16
// ---------------------------------------------------------------------------
__global__ __launch_bounds__(256)
void ln_k(const float* __restrict__ Y, const float* __restrict__ g,
          const float* __restrict__ be, bf16* __restrict__ out) {
  __shared__ float red[256];
  const int r = blockIdx.x;
  const float* y = Y + (size_t)r * DIMC;

  float s = 0.f;
  for (int c = threadIdx.x; c < DIMC; c += 256) s += y[c];
  red[threadIdx.x] = s;
  __syncthreads();
  for (int st = 128; st > 0; st >>= 1) {
    if (threadIdx.x < st) red[threadIdx.x] += red[threadIdx.x + st];
    __syncthreads();
  }
  const float mu = red[0] / (float)DIMC;
  __syncthreads();

  float v = 0.f;
  for (int c = threadIdx.x; c < DIMC; c += 256) {
    float d = y[c] - mu;
    v += d * d;
  }
  red[threadIdx.x] = v;
  __syncthreads();
  for (int st = 128; st > 0; st >>= 1) {
    if (threadIdx.x < st) red[threadIdx.x] += red[threadIdx.x + st];
    __syncthreads();
  }
  const float rstd = rsqrtf(red[0] / (float)DIMC + 1e-5f);
  for (int c = threadIdx.x; c < DIMC; c += 256)
    out[(size_t)r * DIMC + c] = (bf16)((y[c] - mu) * rstd * g[c] + be[c]);
}

// ---------------------------------------------------------------------------
extern "C" void kernel_launch(void* const* d_in, const int* in_sizes, int n_in,
                              void* d_out, int out_size, void* d_ws,
                              size_t ws_size, hipStream_t stream) {
  const float* x   = (const float*)d_in[0];
  const float* h_t = (const float*)d_in[1];
  const float* h_a = (const float*)d_in[2];
  const float* p   = (const float*)d_in[3];
  const float* Wq  = (const float*)d_in[4];
  const float* bq  = (const float*)d_in[5];
  const float* Wk  = (const float*)d_in[6];
  const float* bk  = (const float*)d_in[7];
  const float* Wv  = (const float*)d_in[8];
  const float* bv  = (const float*)d_in[9];
  const float* Wo  = (const float*)d_in[10];
  const float* bo  = (const float*)d_in[11];
  const float* gf  = (const float*)d_in[12];
  const float* lng = (const float*)d_in[13];
  const float* lnb = (const float*)d_in[14];
  const float* Wf  = (const float*)d_in[15];
  const float* bfv = (const float*)d_in[16];
  float* outp      = (float*)d_out;

  char* wsp = (char*)d_ws;
  auto carve = [&](size_t bytes) {
    char* r = wsp;
    wsp += (bytes + 255) & ~(size_t)255;
    return r;
  };
  const size_t WE = (size_t)DIMC * DIMC;  // weight elements
  bf16* WqB = (bf16*)carve(WE * 2);
  bf16* WkB = (bf16*)carve(WE * 2);
  bf16* WvB = (bf16*)carve(WE * 2);
  bf16* WoB = (bf16*)carve(WE * 2);
  bf16* WfB = (bf16*)carve(WE * 2);
  bf16* Xb  = (bf16*)carve((size_t)BB * TT * DIMC * 2);
  bf16* Hc  = (bf16*)carve((size_t)BB * KA * DIMC * 2);
  bf16* Ht  = (bf16*)carve((size_t)BB * KT * DIMC * 2);
  bf16* Qb  = (bf16*)carve((size_t)BB * TT * DIMC * 2);
  bf16* Kb  = (bf16*)carve((size_t)BB * S_ALL * DIMC * 2);
  bf16* Vb  = (bf16*)carve((size_t)BB * S_ALL * DIMC * 2);
  bf16* Ab  = (bf16*)carve((size_t)BB * TT * DIMC * 2);
  // aliases over buffers dead by the time these are written:
  float* Y32 = (float*)Kb;  // post-Wo + residual (32 MB < Kb+Vb span)
  bf16*  Yb  = (bf16*)Qb;   // post-LN bf16

  dim3 blk(256);
  auto cvt = [&](const float* src, bf16* dst, long n_pb, long ostr, long seg,
                 long total) {
    int nb = (int)((total + 255) / 256);
    if (nb > 8192) nb = 8192;
    cvt_bf16_k<<<dim3(nb), blk, 0, stream>>>(src, dst, n_pb, ostr, seg, total);
  };
  cvt(Wq, WqB, (long)WE, (long)WE, 0, (long)WE);
  cvt(Wk, WkB, (long)WE, (long)WE, 0, (long)WE);
  cvt(Wv, WvB, (long)WE, (long)WE, 0, (long)WE);
  cvt(Wo, WoB, (long)WE, (long)WE, 0, (long)WE);
  cvt(Wf, WfB, (long)WE, (long)WE, 0, (long)WE);
  cvt(x,   Xb, (long)BB * TT * DIMC, 0, 0, (long)BB * TT * DIMC);
  cvt(h_t, Ht, (long)BB * KT * DIMC, 0, 0, (long)BB * KT * DIMC);
  cvt(h_a, Hc, 64L * DIMC, (long)KA * DIMC, 0,          (long)BB * 64 * DIMC);
  cvt(p,   Hc, 64L * DIMC, (long)KA * DIMC, 64L * DIMC, (long)BB * 64 * DIMC);

#define GEMM_LAUNCH(MODE, Aq, Wm, biasq, rows, orpb, seg, oB, oF, res)        \
  gemm_bf16_k<MODE><<<dim3(DIMC / 128, (rows) / 128, BB), blk, 0, stream>>>(  \
      Aq, Wm, biasq, rows, orpb, seg, oB, oF, res)

  GEMM_LAUNCH(0, Xb, WqB, bq, TT, TT,    0,         Qb, (float*)nullptr, (const float*)nullptr);
  GEMM_LAUNCH(0, Xb, WkB, bk, TT, S_ALL, 0,         Kb, (float*)nullptr, (const float*)nullptr);
  GEMM_LAUNCH(0, Xb, WvB, bv, TT, S_ALL, 0,         Vb, (float*)nullptr, (const float*)nullptr);
  GEMM_LAUNCH(0, Hc, WkB, bk, KA, S_ALL, TT,        Kb, (float*)nullptr, (const float*)nullptr);
  GEMM_LAUNCH(0, Hc, WvB, bv, KA, S_ALL, TT,        Vb, (float*)nullptr, (const float*)nullptr);
  GEMM_LAUNCH(0, Ht, WkB, bk, KT, S_ALL, S_TSK_OFF, Kb, (float*)nullptr, (const float*)nullptr);
  GEMM_LAUNCH(0, Ht, WvB, bv, KT, S_ALL, S_TSK_OFF, Vb, (float*)nullptr, (const float*)nullptr);

  attn_k<<<dim3(TT / 128, NHEAD, BB), blk, 0, stream>>>(Qb, Kb, Vb, gf, Ab);

  // out = Ab*Wo^T + bo + x   (fp32)
  GEMM_LAUNCH(1, Ab, WoB, bo, TT, TT, 0, (bf16*)nullptr, Y32, x);
  // LayerNorm -> bf16
  ln_k<<<dim3(BB * TT), blk, 0, stream>>>(Y32, lng, lnb, Yb);
  // final: relu(Yb*Wf^T + bf) -> d_out
  GEMM_LAUNCH(2, Yb, WfB, bfv, TT, TT, 0, (bf16*)nullptr, outp, (const float*)nullptr);
#undef GEMM_LAUNCH
}
